// MambaModel2_53352083751354
// MI455X (gfx1250) — matrix operand order repaired and checked by reading, not compiled, not run on hardware
//
#include <hip/hip_runtime.h>
#include <hip/hip_bf16.h>

// ---------------- model dims ----------------
constexpr int cNL = 24;
constexpr int cH  = 768;
constexpr int cI  = 1536;
constexpr int cN  = 16;
constexpr int cR  = 48;
constexpr int cK  = 4;        // conv taps
constexpr int cV  = 50280;
constexpr int cL  = 1024;
constexpr int cB  = 2;
constexpr int cM  = cB * cL;          // 2048 token rows
constexpr int cSS = cR + 2 * cN;      // 80
constexpr int NC  = 8;                // scan chunks
constexpr int CHK = cL / NC;          // 128

// ---------------- WMMA types ----------------
typedef __attribute__((ext_vector_type(16))) __bf16 v16bf;
typedef __attribute__((ext_vector_type(8)))  float  v8f;
typedef unsigned short ushort_t;

__device__ __forceinline__ ushort_t f2bf(float f) {
  return __builtin_bit_cast(ushort_t, (__bf16)f);   // native v_cvt on gfx1250
}
__device__ __forceinline__ unsigned int pack2bf(float a, float b) {
  return (unsigned int)f2bf(a) | ((unsigned int)f2bf(b) << 16);
}
__device__ __forceinline__ __bf16 bfc(ushort_t u) {
  return __builtin_bit_cast(__bf16, u);
}

// ---------------- f32 -> bf16 bulk convert (4 elems/thread) ----------------
__global__ __launch_bounds__(256) void cvt_bf16_kernel(
    const float* __restrict__ in, ushort_t* __restrict__ out, long n4)
{
  long i = (long)blockIdx.x * 256 + threadIdx.x;
  if (i >= n4) return;
  float4 v = reinterpret_cast<const float4*>(in)[i];
  uint2 p; p.x = pack2bf(v.x, v.y); p.y = pack2bf(v.z, v.w);
  reinterpret_cast<uint2*>(out)[i] = p;
}

// ---------------- GEMM: C[M,N] = act(A_bf[M,K] * W_bf[N,K]^T) ----------------
// Block tile 128x128, 8 waves, wave tile 32x64 -> 8 WMMA per k-step.
// M == 2048 always (multiple of 128) -> no M guards.
// modes: 0 plain, 1 softplus(acc+bias[n]), 2 C += acc (residual),
//        3 row-swapped store for logits: row m=b*L+l -> out row l*B+b
// Cbf (optional) gets a bf16 copy of the stored value (used for ssm_in).
#define BM 128
#define BN 128
#define BKK 32

template <bool NEDGE, bool KEDGE>
__global__ __launch_bounds__(256) void gemm_wmma_kernel(
    const ushort_t* __restrict__ A, int lda,
    const ushort_t* __restrict__ W, int ldw,
    const float* __restrict__ bias,
    float* __restrict__ C, ushort_t* __restrict__ Cbf, int ldc,
    int N, int K, int mode, int n0)
{
  __shared__ ushort_t sA[BM * BKK];
  __shared__ ushort_t sB[BN * BKK];

  const int tid  = threadIdx.x;
  const int wave = tid >> 5;      // 0..7
  const int lane = tid & 31;
  const int wm   = wave >> 1;     // 0..3 -> 32-row group
  const int wn   = wave & 1;      // 0..1 -> 64-col group
  const int half = lane >> 4;
  const int l16  = lane & 15;

  const int block_m = blockIdx.y * BM;
  const int block_n = n0 + blockIdx.x * BN;

  const v8f vzero = {0.f, 0.f, 0.f, 0.f, 0.f, 0.f, 0.f, 0.f};
  v8f acc[2][4];
  #pragma unroll
  for (int i = 0; i < 2; ++i)
    #pragma unroll
    for (int j = 0; j < 4; ++j) acc[i][j] = vzero;

  for (int kk = 0; kk < K; kk += BKK) {
    if constexpr (!KEDGE) {
      // pure bf16 copy staging: uint4 (8 bf16) loads -> ds_store_b128
      #pragma unroll
      for (int t = 0; t < (BM * BKK) / (8 * 256); ++t) {   // 2 iters
        int idx = tid + t * 256;          // uint4 index
        int row = idx >> 2, c8 = idx & 3;
        uint4 v = *reinterpret_cast<const uint4*>(
            &A[(long)(block_m + row) * lda + kk + c8 * 8]);
        *reinterpret_cast<uint4*>(&sA[row * BKK + c8 * 8]) = v;
      }
      #pragma unroll
      for (int t = 0; t < (BN * BKK) / (8 * 256); ++t) {
        int idx = tid + t * 256;
        int row = idx >> 2, c8 = idx & 3;
        int gn = block_n + row;
        uint4 v = {0u, 0u, 0u, 0u};
        if (!NEDGE || gn < N)
          v = *reinterpret_cast<const uint4*>(&W[(long)gn * ldw + kk + c8 * 8]);
        *reinterpret_cast<uint4*>(&sB[row * BKK + c8 * 8]) = v;
      }
      if (kk + BKK < K) {
        __builtin_prefetch(&A[(long)(block_m + (tid >> 1)) * lda + kk + BKK], 0, 1);
        __builtin_prefetch(&W[(long)(block_n + (tid >> 1)) * ldw + kk + BKK], 0, 1);
      }
    } else {
      // guarded scalar staging (K edge; N guard folds away if !NEDGE)
      #pragma unroll
      for (int t = 0; t < (BM * BKK) / 256; ++t) {
        int idx = tid + t * 256;
        int r = idx >> 5, c = idx & 31;
        int gk = kk + c;
        sA[idx] = (gk < K) ? A[(long)(block_m + r) * lda + gk] : (ushort_t)0;
      }
      #pragma unroll
      for (int t = 0; t < (BN * BKK) / 256; ++t) {
        int idx = tid + t * 256;
        int r = idx >> 5, c = idx & 31;
        int gn = block_n + r, gk = kk + c;
        sB[idx] = ((!NEDGE || gn < N) && gk < K) ? W[(long)gn * ldw + gk]
                                                 : (ushort_t)0;
      }
    }
    __syncthreads();

    // A fragments: 16x32 bf16 layout (ISA 7.12.2)
    v16bf afrag[2], bfrag[4];
    #pragma unroll
    for (int i = 0; i < 2; ++i) {
      int mrow = wm * 32 + i * 16 + l16;
      const ushort_t* pa = &sA[mrow * BKK];
      #pragma unroll
      for (int g = 0; g < 8; ++g) {
        int k0 = ((g < 4) ? (2 * g) : (16 + 2 * (g - 4))) + half * 8;
        afrag[i][2 * g]     = bfc(pa[k0]);
        afrag[i][2 * g + 1] = bfc(pa[k0 + 1]);
      }
    }
    // B fragments: lane holds column n, K order same as A fragments
    #pragma unroll
    for (int j = 0; j < 4; ++j) {
      int ncol = wn * 64 + j * 16 + l16;
      const ushort_t* pb = &sB[ncol * BKK];
      #pragma unroll
      for (int g = 0; g < 8; ++g) {
        int k0 = ((g < 4) ? (2 * g) : (16 + 2 * (g - 4))) + half * 8;
        bfrag[j][2 * g]     = bfc(pb[k0]);
        bfrag[j][2 * g + 1] = bfc(pb[k0 + 1]);
      }
    }

    #pragma unroll
    for (int i = 0; i < 2; ++i)
      #pragma unroll
      for (int j = 0; j < 4; ++j)
        acc[i][j] = __builtin_amdgcn_wmma_f32_16x16x32_bf16(
            false, afrag[i], false, bfrag[j], (short)0, acc[i][j], false, false);
    __syncthreads();
  }

  // epilogue: C/D layout = 8 VGPRs; VGPR r: row = r + 8*half, col = lane&15
  #pragma unroll
  for (int i = 0; i < 2; ++i) {
    #pragma unroll
    for (int j = 0; j < 4; ++j) {
      int tile_m = block_m + wm * 32 + i * 16;
      int col = block_n + wn * 64 + j * 16 + l16;
      if (NEDGE && col >= N) continue;
      #pragma unroll
      for (int r = 0; r < 8; ++r) {
        int row = tile_m + r + half * 8;
        float v = acc[i][j][r];
        if (mode == 1) {                       // softplus(acc + bdt)
          v += bias[col];
          v = (v > 20.f) ? v : log1pf(__expf(v));
        }
        long off;
        if (mode == 3) {                       // (b*L+l) -> (l*B+b)
          int bb = row >> 10, ll = row & (cL - 1);
          off = ((long)ll * cB + bb) * ldc + col;
        } else {
          off = (long)row * ldc + col;
        }
        if (mode == 2) C[off] += v; else C[off] = v;
        if (Cbf) Cbf[off] = f2bf(v);
      }
    }
  }
}

// ---------------- embedding gather ----------------
__global__ __launch_bounds__(256) void embed_kernel(
    const int* __restrict__ src, const float* __restrict__ emb,
    float* __restrict__ h)
{
  int m = blockIdx.x;                 // m = b*L + l
  int b = m / cL, l = m % cL;
  int tok = src[l * cB + b];          // src is (L,B)
  const float* e = emb + (long)tok * cH;
  float* hr = h + (long)m * cH;
  for (int c = threadIdx.x; c < cH; c += 256) hr[c] = e[c];
}

// ---------------- RMSNorm -> bf16 ----------------
__global__ __launch_bounds__(256) void rmsnorm_kernel(
    const float* __restrict__ x, const float* __restrict__ w,
    ushort_t* __restrict__ out)
{
  __shared__ float red[256];
  int m = blockIdx.x;
  const float* xr = x + (long)m * cH;
  float s = 0.f;
  for (int c = threadIdx.x; c < cH; c += 256) { float v = xr[c]; s += v * v; }
  red[threadIdx.x] = s;
  __syncthreads();
  for (int off = 128; off > 0; off >>= 1) {
    if (threadIdx.x < off) red[threadIdx.x] += red[threadIdx.x + off];
    __syncthreads();
  }
  float inv = rsqrtf(red[0] / (float)cH + 1e-5f);
  for (int c = threadIdx.x; c < cH; c += 256)
    out[(long)m * cH + c] = f2bf(xr[c] * inv * w[c]);
}

// ---------------- causal depthwise conv (K=4) + SiLU ----------------
__global__ __launch_bounds__(256) void conv_silu_kernel(
    const float* __restrict__ xz, const float* __restrict__ cw,
    const float* __restrict__ cb, float* __restrict__ xs,
    ushort_t* __restrict__ xs_bf)
{
  long idx = (long)blockIdx.x * 256 + threadIdx.x;   // over M*I
  int i = (int)(idx % cI);
  long m = idx / cI;                                  // b*L + l
  int l = (int)(m % cL);
  float acc = cb[i];
  #pragma unroll
  for (int k = 0; k < cK; ++k) {
    int dl = l - (cK - 1) + k;
    if (dl >= 0) acc += xz[(m - (cK - 1) + k) * (2 * cI) + i] * cw[i * cK + k];
  }
  float s = acc / (1.f + __expf(-acc));
  xs[idx]    = s;
  xs_bf[idx] = f2bf(s);
}

// ---------------- chunked SSM scan ----------------
// h_l = dA_l * h_{l-1} + dBx_l (diagonal per (i,n)); y_l = <h_l, C_l> + D*xs
// Pass A: per (b,chunk,i): P = prod(dA), S = chunk scan from 0.
// Pass B: per (b,i): serially combine 8 chunks, record initial states.
// Pass C: per (b,chunk,i): rescan chunk from its initial state, emit y (bf16).
__global__ __launch_bounds__(256) void scan_partial_kernel(
    const float* __restrict__ dt, const float* __restrict__ xs,
    const float* __restrict__ ssm, const float* __restrict__ A_log,
    float* __restrict__ wsP, float* __restrict__ wsS)
{
  int idx = blockIdx.x * 256 + threadIdx.x;   // 0 .. B*NC*I-1
  int i = idx % cI;
  int t = idx / cI;
  int c = t % NC, b = t / NC;
  float a[cN], P[cN], S[cN];
  #pragma unroll
  for (int n = 0; n < cN; ++n) {
    a[n] = -__expf(A_log[i * cN + n]); P[n] = 1.f; S[n] = 0.f;
  }
  for (int l = c * CHK; l < (c + 1) * CHK; ++l) {
    long m = (long)b * cL + l;
    float dtv = dt[m * cI + i];
    float xsv = xs[m * cI + i];
    const float* Bm = &ssm[m * cSS + cR];
    float dtxs = dtv * xsv;
    #pragma unroll
    for (int n = 0; n < cN; ++n) {
      float dA = __expf(dtv * a[n]);
      S[n] = dA * S[n] + dtxs * Bm[n];
      P[n] *= dA;
    }
  }
  #pragma unroll
  for (int n = 0; n < cN; ++n) {
    wsP[(long)idx * cN + n] = P[n];
    wsS[(long)idx * cN + n] = S[n];
  }
}

__global__ __launch_bounds__(256) void scan_combine_kernel(
    const float* __restrict__ wsP, const float* __restrict__ wsS,
    float* __restrict__ wsInit)
{
  int idx = blockIdx.x * 256 + threadIdx.x;   // 0 .. B*I-1
  int i = idx % cI, b = idx / cI;
  float H[cN];
  #pragma unroll
  for (int n = 0; n < cN; ++n) H[n] = 0.f;
  for (int c = 0; c < NC; ++c) {
    long rec = ((long)(b * NC + c) * cI + i) * cN;
    #pragma unroll
    for (int n = 0; n < cN; ++n) {
      wsInit[rec + n] = H[n];
      H[n] = wsP[rec + n] * H[n] + wsS[rec + n];
    }
  }
}

__global__ __launch_bounds__(256) void scan_final_kernel(
    const float* __restrict__ dt, const float* __restrict__ xs,
    const float* __restrict__ ssm, const float* __restrict__ xz,
    const float* __restrict__ A_log, const float* __restrict__ Dp,
    const float* __restrict__ wsInit, ushort_t* __restrict__ y_bf)
{
  int idx = blockIdx.x * 256 + threadIdx.x;   // 0 .. B*NC*I-1
  int i = idx % cI;
  int t = idx / cI;
  int c = t % NC, b = t / NC;
  float a[cN], hs[cN];
  #pragma unroll
  for (int n = 0; n < cN; ++n) {
    a[n] = -__expf(A_log[i * cN + n]);
    hs[n] = wsInit[(long)idx * cN + n];
  }
  float Dv = Dp[i];
  for (int l = c * CHK; l < (c + 1) * CHK; ++l) {
    long m = (long)b * cL + l;
    float dtv = dt[m * cI + i];
    float xsv = xs[m * cI + i];
    float zv  = xz[m * (2 * cI) + cI + i];
    const float* Bm = &ssm[m * cSS + cR];
    const float* Cm = &ssm[m * cSS + cR + cN];
    float dtxs = dtv * xsv;
    float yv = 0.f;
    #pragma unroll
    for (int n = 0; n < cN; ++n) {
      float dA = __expf(dtv * a[n]);
      hs[n] = dA * hs[n] + dtxs * Bm[n];
      yv += hs[n] * Cm[n];
    }
    yv += Dv * xsv;
    float sz = zv / (1.f + __expf(-zv));      // silu(z)
    y_bf[m * cI + i] = f2bf(yv * sz);
  }
}

// ---------------- driver ----------------
static inline void cvt_launch(const float* in, ushort_t* out, long n,
                              hipStream_t stream) {
  long n4 = n / 4;
  cvt_bf16_kernel<<<(unsigned)((n4 + 255) / 256), 256, 0, stream>>>(in, out, n4);
}

extern "C" void kernel_launch(void* const* d_in, const int* in_sizes, int n_in,
                              void* d_out, int out_size, void* d_ws, size_t ws_size,
                              hipStream_t stream) {
  const int*   src    = (const int*)  d_in[0];
  const float* norm_w = (const float*)d_in[1];
  const float* Win    = (const float*)d_in[2];
  const float* conv_w = (const float*)d_in[3];
  const float* conv_b = (const float*)d_in[4];
  const float* Wx     = (const float*)d_in[5];
  const float* Wdt    = (const float*)d_in[6];
  const float* bdt    = (const float*)d_in[7];
  const float* A_log  = (const float*)d_in[8];
  const float* Dp     = (const float*)d_in[9];
  const float* Wout   = (const float*)d_in[10];
  const float* norm_f = (const float*)d_in[11];
  const float* emb    = (const float*)d_in[12];
  float* logits = (float*)d_out;

  float* ws = (float*)d_ws;
  float* h     = ws;  ws += (long)cM * cH;
  float* xz    = ws;  ws += (long)cM * 2 * cI;
  float* xs    = ws;  ws += (long)cM * cI;
  float* ssm   = ws;  ws += (long)cM * cSS;
  float* dt    = ws;  ws += (long)cM * cI;
  float* wsP   = ws;  ws += (long)cB * NC * cI * cN;
  float* wsS   = ws;  ws += (long)cB * NC * cI * cN;
  float* wsIni = ws;  ws += (long)cB * NC * cI * cN;
  // bf16 buffers (ushort), sizes all multiples of 8 -> 16B alignment kept
  ushort_t* xn_bf  = (ushort_t*)ws;  ws += (long)cM * cH / 2;
  ushort_t* xs_bf  = (ushort_t*)ws;  ws += (long)cM * cI / 2;
  ushort_t* ssm_bf = (ushort_t*)ws;  ws += (long)cM * cSS / 2;
  ushort_t* y_bf   = (ushort_t*)ws;  ws += (long)cM * cI / 2;
  ushort_t* Win_bf = (ushort_t*)ws;  ws += (long)2 * cI * cH / 2;
  ushort_t* Wx_bf  = (ushort_t*)ws;  ws += (long)cSS * cI / 2;
  ushort_t* Wdt_bf = (ushort_t*)ws;  ws += (long)cI * cR / 2;
  ushort_t* Wout_bf= (ushort_t*)ws;  ws += (long)cH * cI / 2;
  ushort_t* emb_bf = (ushort_t*)ws;  ws += (long)cV * cH / 2;

  embed_kernel<<<cM, 256, 0, stream>>>(src, emb, h);

  const dim3 blk(256);
  const int mTiles = cM / BM;                   // 16
  for (int l = 0; l < cNL; ++l) {
    // convert this layer's weights to bf16 (once, reused by all tiles)
    cvt_launch(Win  + (long)l * 2 * cI * cH, Win_bf,  (long)2 * cI * cH, stream);
    cvt_launch(Wx   + (long)l * cSS * cI,    Wx_bf,   (long)cSS * cI,    stream);
    cvt_launch(Wdt  + (long)l * cI * cR,     Wdt_bf,  (long)cI * cR,     stream);
    cvt_launch(Wout + (long)l * cH * cI,     Wout_bf, (long)cH * cI,     stream);

    rmsnorm_kernel<<<cM, 256, 0, stream>>>(h, norm_w + (long)l * cH, xn_bf);

    // xz = xn @ Win^T : N=3072, K=768 (fully aligned)
    gemm_wmma_kernel<false, false><<<dim3(2 * cI / BN, mTiles), blk, 0, stream>>>(
        xn_bf, cH, Win_bf, cH, nullptr, xz, nullptr, 2 * cI, 2 * cI, cH, 0, 0);

    conv_silu_kernel<<<((long)cM * cI) / 256, blk, 0, stream>>>(
        xz, conv_w + (long)l * cI * cK, conv_b + (long)l * cI, xs, xs_bf);

    // ssm_in = xs @ Wx^T : N=80 (edge), K=1536; dual-write f32 + bf16
    gemm_wmma_kernel<true, false><<<dim3(1, mTiles), blk, 0, stream>>>(
        xs_bf, cI, Wx_bf, cI, nullptr, ssm, ssm_bf, cSS, cSS, cI, 0, 0);

    // dt = softplus(dt_r @ Wdt^T + bdt) : N=1536, K=48 (edge)
    gemm_wmma_kernel<false, true><<<dim3(cI / BN, mTiles), blk, 0, stream>>>(
        ssm_bf, cSS, Wdt_bf, cR, bdt + (long)l * cI, dt, nullptr, cI, cI, cR, 1, 0);

    // chunked scan
    scan_partial_kernel<<<(cB * NC * cI) / 256, blk, 0, stream>>>(
        dt, xs, ssm, A_log + (long)l * cI * cN, wsP, wsS);
    scan_combine_kernel<<<(cB * cI) / 256, blk, 0, stream>>>(wsP, wsS, wsIni);
    scan_final_kernel<<<(cB * NC * cI) / 256, blk, 0, stream>>>(
        dt, xs, ssm, xz, A_log + (long)l * cI * cN, Dp + (long)l * cI, wsIni, y_bf);

    // h += y @ Wout^T : N=768, K=1536 (aligned, residual accumulate)
    gemm_wmma_kernel<false, false><<<dim3(cH / BN, mTiles), blk, 0, stream>>>(
        y_bf, cI, Wout_bf, cI, nullptr, h, nullptr, cH, cH, cI, 2, 0);
  }

  rmsnorm_kernel<<<cM, 256, 0, stream>>>(h, norm_f, xn_bf);

  // logits = xn @ emb^T : body (392 aligned tiles) + tail (104 cols)
  cvt_launch(emb, emb_bf, (long)cV * cH, stream);
  constexpr int vBody = (cV / BN) * BN;         // 50176
  gemm_wmma_kernel<false, false><<<dim3(cV / BN, mTiles), blk, 0, stream>>>(
      xn_bf, cH, emb_bf, cH, nullptr, logits, nullptr, cV, cV, cH, 3, 0);
  gemm_wmma_kernel<true, false><<<dim3(1, mTiles), blk, 0, stream>>>(
      xn_bf, cH, emb_bf, cH, nullptr, logits, nullptr, cV, cV, cH, 3, vBody);
}
